// FABiS6Block_45165876085135
// MI455X (gfx1250) — matrix operand order repaired and checked by reading, not compiled, not run on hardware
//
#include <hip/hip_runtime.h>
#include <hip/hip_bf16.h>
#include <math.h>

#define S_LEN 1024
#define BATCH 8
#define DMODEL 768
#define NROWS (S_LEN * BATCH)      /* 8192 */
#define CS 16
#define NCHUNK (S_LEN / CS)        /* 64 */
#define NSTATE (NCHUNK * BATCH)    /* 512 */
#define HELEMS (NSTATE * DMODEL)   /* 393216 */

typedef __attribute__((ext_vector_type(16))) __bf16 v16bf;
typedef __attribute__((ext_vector_type(8)))  float  v8f;

union FragBF {
  v16bf v;
  uint4 q[2];
  unsigned short u[16];
};

__device__ __forceinline__ unsigned short f2bf(float f) {
  unsigned int u = __float_as_uint(f);
  u += 0x7fffu + ((u >> 16) & 1u);   // round to nearest even
  return (unsigned short)(u >> 16);
}

// A fragment (16x32 bf16, M rows x K): lane L holds row rows[L%16];
// elements 0..7 = K k0+8*hi .. +7, elements 8..15 = K k0+16+8*hi .. +7  (hi = L/16)
__device__ __forceinline__ v16bf load_frag_a(const unsigned short* row_ptr, int k0, int hi) {
  FragBF f;
  const unsigned short* p = row_ptr + k0 + hi * 8;
  f.q[0] = *(const uint4*)(p);
  f.q[1] = *(const uint4*)(p + 16);
  return f.v;
}

// B fragment (32x16 bf16, K x N) built from row-major W (N rows, K cols):
// lane L holds column n = n_abs (passed per-lane), elements 0..7 = K k0+8*hi .. +7, elements 8..15 = K k0+16+8*hi .. +7.
__device__ __forceinline__ v16bf load_frag_b(const unsigned short* w, int n, int k0, int hi) {
  FragBF f;
  const unsigned short* p = w + (long)n * DMODEL + k0 + hi * 8;
  f.q[0] = *(const uint4*)(p);
  f.q[1] = *(const uint4*)(p + 16);
  return f.v;
}

__device__ __forceinline__ v16bf zero_frag() {
  FragBF f;
#pragma unroll
  for (int i = 0; i < 16; ++i) f.u[i] = 0;
  return f.v;
}

__device__ __forceinline__ v8f wmma_bf16(v16bf a, v16bf b, v8f c) {
  return __builtin_amdgcn_wmma_f32_16x16x32_bf16(false, a, false, b, (short)0, c, false, false);
}

__device__ __forceinline__ float block_sum256(float v) {
  __shared__ float red[256];
  int t = threadIdx.x;
  red[t] = v;
  __syncthreads();
  for (int s = 128; s > 0; s >>= 1) {
    if (t < s) red[t] += red[t + s];
    __syncthreads();
  }
  float r = red[0];
  __syncthreads();
  return r;
}

// ---------------- conversion / packing ----------------

__global__ void cvt_f32_bf16(const float* __restrict__ in, unsigned short* __restrict__ out, int n) {
  int i = blockIdx.x * 256 + threadIdx.x;
  if (i < n) out[i] = f2bf(in[i]);
}

// (oc=256, D=768, k) -> tap-major bf16 [j][oc][D]
__global__ void pack_conv(const float* __restrict__ w, unsigned short* __restrict__ out,
                          int k, int total) {
  int idx = blockIdx.x * 256 + threadIdx.x;
  if (idx >= total) return;
  int o   = idx / (DMODEL * k);
  int rem = idx - o * DMODEL * k;
  int i   = rem / k;
  int j   = rem - i * k;
  out[(long)j * 256 * DMODEL + (long)o * DMODEL + i] = f2bf(w[idx]);
}

// ---------------- conv branch GEMM (all 6 convs) ----------------
// grid (512 M-tiles, 6 convs), block 128 (4 waves); wave -> 16x64 tile, 4 N-subtiles.
__global__ void conv_gemm(const unsigned short* __restrict__ xb,
                          const unsigned short* __restrict__ wconv,
                          const float* __restrict__ b0, const float* __restrict__ b1,
                          const float* __restrict__ b2, const float* __restrict__ b3,
                          const float* __restrict__ b4, const float* __restrict__ b5,
                          float* __restrict__ tfa_raw, float* __restrict__ cfa_raw) {
  const int ktab[6]  = {2, 3, 4, 2, 4, 8};
  const int otab[6]  = {-1, -1, -2, -1, -2, -4};
  const int wbase[6] = {0, 2, 5, 9, 11, 15};    // cumulative taps

  int mtile = blockIdx.x;
  int c     = blockIdx.y;
  int lane  = threadIdx.x & 31;
  int wv    = threadIdx.x >> 5;
  int hi    = lane >> 4;
  int ml    = lane & 15;

  int k = ktab[c], off = otab[c];
  const unsigned short* wp = wconv + (long)wbase[c] * 256 * DMODEL;
  const float* bias = (c == 0) ? b0 : (c == 1) ? b1 : (c == 2) ? b2
                    : (c == 3) ? b3 : (c == 4) ? b4 : b5;
  float* out = (c < 3) ? tfa_raw : cfa_raw;
  int sub = c % 3;
  int arow = mtile * 16 + ml;

  v8f acc[4];
#pragma unroll
  for (int ns = 0; ns < 4; ++ns)
    for (int i = 0; i < 8; ++i) acc[ns][i] = 0.f;

  for (int j = 0; j < k; ++j) {
    int shrow = arow + (off + j) * BATCH;
    bool valid = (shrow >= 0) && (shrow < NROWS);
    const unsigned short* ap = xb + (long)shrow * DMODEL;
    const unsigned short* wj = wp + (long)j * 256 * DMODEL;
    for (int k0 = 0; k0 < DMODEL; k0 += 32) {
      v16bf a = valid ? load_frag_a(ap, k0, hi) : zero_frag();
#pragma unroll
      for (int ns = 0; ns < 4; ++ns) {
        int n = wv * 64 + ns * 16 + ml;
        v16bf b = load_frag_b(wj, n, k0, hi);
        acc[ns] = wmma_bf16(a, b, acc[ns]);
      }
    }
  }
#pragma unroll
  for (int ns = 0; ns < 4; ++ns) {
    int ncol = wv * 64 + ns * 16 + ml;
    float bv = bias[ncol];
    int col = sub * 256 + ncol;
#pragma unroll
    for (int v = 0; v < 8; ++v) {
      int m = v + 8 * hi;
      long row = (long)(mtile * 16 + m);
      out[row * DMODEL + col] = acc[ns][v] + bv;
    }
  }
}

// ---------------- LN(tfa)+LN(cfa) -> LN(sum) = agg ----------------
__global__ void ln_agg_kernel(const float* __restrict__ tfa, const float* __restrict__ cfa,
                              const float* __restrict__ tg, const float* __restrict__ tb,
                              const float* __restrict__ cg, const float* __restrict__ cb,
                              const float* __restrict__ fg, const float* __restrict__ fb,
                              float* __restrict__ aggf, unsigned short* __restrict__ aggbf) {
  long r = blockIdx.x;
  int t = threadIdx.x;
  const float inv = 1.0f / (float)DMODEL, eps = 1e-5f;
  float tv[3], cv[3];
  float st = 0.f, sc = 0.f;
#pragma unroll
  for (int i = 0; i < 3; ++i) {
    int d = t + i * 256;
    tv[i] = tfa[r * DMODEL + d];
    cv[i] = cfa[r * DMODEL + d];
    st += tv[i]; sc += cv[i];
  }
  float mt = block_sum256(st) * inv;
  float mc = block_sum256(sc) * inv;
  float vt = 0.f, vc = 0.f;
#pragma unroll
  for (int i = 0; i < 3; ++i) {
    float a = tv[i] - mt; vt += a * a;
    float b = cv[i] - mc; vc += b * b;
  }
  vt = block_sum256(vt) * inv;
  vc = block_sum256(vc) * inv;
  float rt = 1.0f / sqrtf(vt + eps);
  float rc = 1.0f / sqrtf(vc + eps);
  float sv[3]; float ss = 0.f;
#pragma unroll
  for (int i = 0; i < 3; ++i) {
    int d = t + i * 256;
    sv[i] = ((tv[i] - mt) * rt) * tg[d] + tb[d] + ((cv[i] - mc) * rc) * cg[d] + cb[d];
    ss += sv[i];
  }
  float ms = block_sum256(ss) * inv;
  float vs = 0.f;
#pragma unroll
  for (int i = 0; i < 3; ++i) { float a = sv[i] - ms; vs += a * a; }
  vs = block_sum256(vs) * inv;
  float rs = 1.0f / sqrtf(vs + eps);
#pragma unroll
  for (int i = 0; i < 3; ++i) {
    int d = t + i * 256;
    float o = ((sv[i] - ms) * rs) * fg[d] + fb[d];
    aggf[r * DMODEL + d] = o;
    aggbf[r * DMODEL + d] = f2bf(o);
  }
}

// ---------------- selectivity GEMM: sel = sigmoid(agg @ sw^T + sb) ----------------
// grid (512, 3), block 128
__global__ void sel_gemm(const unsigned short* __restrict__ aggbf,
                         const unsigned short* __restrict__ wsw,
                         const float* __restrict__ sb, float* __restrict__ sel) {
  int mtile = blockIdx.x, cg = blockIdx.y;
  int lane = threadIdx.x & 31, wv = threadIdx.x >> 5;
  int hi = lane >> 4, ml = lane & 15;
  int row = mtile * 16 + ml;
  const unsigned short* ap = aggbf + (long)row * DMODEL;

  v8f acc[4];
#pragma unroll
  for (int ns = 0; ns < 4; ++ns)
    for (int i = 0; i < 8; ++i) acc[ns][i] = 0.f;

  for (int k0 = 0; k0 < DMODEL; k0 += 32) {
    v16bf a = load_frag_a(ap, k0, hi);
#pragma unroll
    for (int ns = 0; ns < 4; ++ns) {
      int n = cg * 256 + wv * 64 + ns * 16 + ml;
      acc[ns] = wmma_bf16(a, load_frag_b(wsw, n, k0, hi), acc[ns]);
    }
  }
#pragma unroll
  for (int ns = 0; ns < 4; ++ns) {
    int col = cg * 256 + wv * 64 + ns * 16 + ml;
    float bv = sb[col];
#pragma unroll
    for (int v = 0; v < 8; ++v) {
      int m = v + 8 * hi;
      long r = (long)(mtile * 16 + m);
      float x = acc[ns][v] + bv;
      sel[r * DMODEL + col] = 1.0f / (1.0f + expf(-x));
    }
  }
}

// ---------------- S6 step A: H = tanh(X_t A^T + Hprev Bm^T) * sel ----------------
// grid (32, 3), block 128; 512 state rows = (chunk, batch)
__global__ void s6_stepA(const unsigned short* __restrict__ aggbf,
                         const unsigned short* __restrict__ hin,
                         unsigned short* __restrict__ hout,
                         const float* __restrict__ sel,
                         const unsigned short* __restrict__ wA,
                         const unsigned short* __restrict__ wBm,
                         int t, int dir, int useH) {
  int mtile = blockIdx.x, cg = blockIdx.y;
  int lane = threadIdx.x & 31, wv = threadIdx.x >> 5;
  int hi = lane >> 4, ml = lane & 15;

  int sr = mtile * 16 + ml;                 // state row for A-lane
  int cc = sr >> 3, bb = sr & 7;
  int pos = cc * CS + t;
  int srow = dir ? (S_LEN - 1 - pos) : pos;
  long xr = (long)srow * BATCH + bb;
  const unsigned short* xp = aggbf + xr * DMODEL;
  const unsigned short* hp = hin + (long)sr * DMODEL;

  v8f acc[4];
#pragma unroll
  for (int ns = 0; ns < 4; ++ns)
    for (int i = 0; i < 8; ++i) acc[ns][i] = 0.f;

  for (int k0 = 0; k0 < DMODEL; k0 += 32) {
    v16bf a = load_frag_a(xp, k0, hi);
#pragma unroll
    for (int ns = 0; ns < 4; ++ns) {
      int n = cg * 256 + wv * 64 + ns * 16 + ml;
      acc[ns] = wmma_bf16(a, load_frag_b(wA, n, k0, hi), acc[ns]);
    }
  }
  if (useH) {
    for (int k0 = 0; k0 < DMODEL; k0 += 32) {
      v16bf a = load_frag_a(hp, k0, hi);
#pragma unroll
      for (int ns = 0; ns < 4; ++ns) {
        int n = cg * 256 + wv * 64 + ns * 16 + ml;
        acc[ns] = wmma_bf16(a, load_frag_b(wBm, n, k0, hi), acc[ns]);
      }
    }
  }
#pragma unroll
  for (int ns = 0; ns < 4; ++ns) {
    int col = cg * 256 + wv * 64 + ns * 16 + ml;
#pragma unroll
    for (int v = 0; v < 8; ++v) {
      int m = v + 8 * hi;
      int sr2 = mtile * 16 + m;
      int c2 = sr2 >> 3, b2 = sr2 & 7;
      int pos2 = c2 * CS + t;
      int srow2 = dir ? (S_LEN - 1 - pos2) : pos2;
      long xr2 = (long)srow2 * BATCH + b2;
      float sv = sel[xr2 * DMODEL + col];
      float hv = tanhf(acc[ns][v]) * sv;
      hout[(long)sr2 * DMODEL + col] = f2bf(hv);
    }
  }
}

// ---------------- S6 step B: Y = H C^T + X_t * Dv ----------------
__global__ void s6_stepB(const unsigned short* __restrict__ hbuf,
                         const float* __restrict__ aggf,
                         const unsigned short* __restrict__ wC,
                         const float* __restrict__ Dv,
                         float* __restrict__ ys, int t, int dir) {
  int mtile = blockIdx.x, cg = blockIdx.y;
  int lane = threadIdx.x & 31, wv = threadIdx.x >> 5;
  int hi = lane >> 4, ml = lane & 15;

  int sr = mtile * 16 + ml;
  const unsigned short* hp = hbuf + (long)sr * DMODEL;

  v8f acc[4];
#pragma unroll
  for (int ns = 0; ns < 4; ++ns)
    for (int i = 0; i < 8; ++i) acc[ns][i] = 0.f;

  for (int k0 = 0; k0 < DMODEL; k0 += 32) {
    v16bf a = load_frag_a(hp, k0, hi);
#pragma unroll
    for (int ns = 0; ns < 4; ++ns) {
      int n = cg * 256 + wv * 64 + ns * 16 + ml;
      acc[ns] = wmma_bf16(a, load_frag_b(wC, n, k0, hi), acc[ns]);
    }
  }
#pragma unroll
  for (int ns = 0; ns < 4; ++ns) {
    int col = cg * 256 + wv * 64 + ns * 16 + ml;
    float dv = Dv[col];
#pragma unroll
    for (int v = 0; v < 8; ++v) {
      int m = v + 8 * hi;
      int sr2 = mtile * 16 + m;
      int c2 = sr2 >> 3, b2 = sr2 & 7;
      int pos2 = c2 * CS + t;
      int srow2 = dir ? (S_LEN - 1 - pos2) : pos2;
      long xr2 = (long)srow2 * BATCH + b2;
      ys[xr2 * DMODEL + col] = acc[ns][v] + aggf[xr2 * DMODEL + col] * dv;
    }
  }
}

// ---------------- final: LN(fwd), LN(bwd), LN(concat 1536) ----------------
__global__ void final_ln_kernel(const float* __restrict__ ysf, const float* __restrict__ ysb,
                                const float* __restrict__ ng, const float* __restrict__ nb,
                                const float* __restrict__ fg, const float* __restrict__ fb,
                                float* __restrict__ out) {
  long r = blockIdx.x;
  int t = threadIdx.x;
  const float inv = 1.0f / (float)DMODEL, eps = 1e-5f;
  float fv[3], bv[3];
  float sf = 0.f, sb2 = 0.f;
#pragma unroll
  for (int i = 0; i < 3; ++i) {
    int d = t + i * 256;
    fv[i] = ysf[r * DMODEL + d];
    bv[i] = ysb[r * DMODEL + d];
    sf += fv[i]; sb2 += bv[i];
  }
  float mf = block_sum256(sf) * inv;
  float mb = block_sum256(sb2) * inv;
  float vf = 0.f, vb = 0.f;
#pragma unroll
  for (int i = 0; i < 3; ++i) {
    float a = fv[i] - mf; vf += a * a;
    float b = bv[i] - mb; vb += b * b;
  }
  vf = block_sum256(vf) * inv;
  vb = block_sum256(vb) * inv;
  float rf = 1.0f / sqrtf(vf + eps);
  float rb = 1.0f / sqrtf(vb + eps);
  float s2 = 0.f;
#pragma unroll
  for (int i = 0; i < 3; ++i) {
    int d = t + i * 256;
    fv[i] = ((fv[i] - mf) * rf) * ng[d] + nb[d];
    bv[i] = ((bv[i] - mb) * rb) * ng[DMODEL + d] + nb[DMODEL + d];
    s2 += fv[i] + bv[i];
  }
  const float inv2 = 1.0f / (float)(2 * DMODEL);
  float m2 = block_sum256(s2) * inv2;
  float v2 = 0.f;
#pragma unroll
  for (int i = 0; i < 3; ++i) {
    float a = fv[i] - m2; v2 += a * a;
    float b = bv[i] - m2; v2 += b * b;
  }
  v2 = block_sum256(v2) * inv2;
  float r2 = 1.0f / sqrtf(v2 + eps);
#pragma unroll
  for (int i = 0; i < 3; ++i) {
    int d = t + i * 256;
    out[r * (2 * DMODEL) + d]          = ((fv[i] - m2) * r2) * fg[d] + fb[d];
    out[r * (2 * DMODEL) + DMODEL + d] = ((bv[i] - m2) * r2) * fg[DMODEL + d] + fb[DMODEL + d];
  }
}

// ---------------- host ----------------
extern "C" void kernel_launch(void* const* d_in, const int* in_sizes, int n_in,
                              void* d_out, int out_size, void* d_ws, size_t ws_size,
                              hipStream_t stream) {
  (void)in_sizes; (void)n_in; (void)out_size; (void)ws_size;

  const float* x = (const float*)d_in[0];
  const float* cw[6] = {(const float*)d_in[1], (const float*)d_in[3], (const float*)d_in[5],
                        (const float*)d_in[7], (const float*)d_in[9], (const float*)d_in[11]};
  const float* cb[6] = {(const float*)d_in[2], (const float*)d_in[4], (const float*)d_in[6],
                        (const float*)d_in[8], (const float*)d_in[10], (const float*)d_in[12]};
  const float* tfa_g   = (const float*)d_in[13];
  const float* cfa_g   = (const float*)d_in[14];
  const float* feat_g  = (const float*)d_in[15];
  const float* tfa_be  = (const float*)d_in[16];
  const float* cfa_be  = (const float*)d_in[17];
  const float* feat_be = (const float*)d_in[18];
  const float* s6A  = (const float*)d_in[19];
  const float* s6Bm = (const float*)d_in[20];
  const float* s6C  = (const float*)d_in[21];
  const float* s6Dv = (const float*)d_in[22];
  const float* s6sw = (const float*)d_in[23];
  const float* s6sb = (const float*)d_in[24];
  const float* s6ng = (const float*)d_in[25];
  const float* s6nb = (const float*)d_in[26];
  const float* fing = (const float*)d_in[27];
  const float* finb = (const float*)d_in[28];

  char* p = (char*)d_ws;
  auto take = [&](size_t bytes) -> char* {
    char* r = p;
    p += (bytes + 255) & ~(size_t)255;
    return r;
  };
  unsigned short* xb    = (unsigned short*)take((size_t)NROWS * DMODEL * 2);
  unsigned short* wconv = (unsigned short*)take((size_t)23 * 256 * DMODEL * 2);
  unsigned short* wA    = (unsigned short*)take((size_t)2 * DMODEL * DMODEL * 2);
  unsigned short* wBm   = (unsigned short*)take((size_t)2 * DMODEL * DMODEL * 2);
  unsigned short* wC    = (unsigned short*)take((size_t)2 * DMODEL * DMODEL * 2);
  unsigned short* wsw   = (unsigned short*)take((size_t)2 * DMODEL * DMODEL * 2);
  float* tfa_raw        = (float*)take((size_t)NROWS * DMODEL * 4);  // later reused as ys_fwd
  float* cfa_raw        = (float*)take((size_t)NROWS * DMODEL * 4);  // later reused as ys_bwd
  float* aggf           = (float*)take((size_t)NROWS * DMODEL * 4);
  unsigned short* aggbf = (unsigned short*)take((size_t)NROWS * DMODEL * 2);
  float* sel            = (float*)take((size_t)NROWS * DMODEL * 4);
  unsigned short* hbf   = (unsigned short*)take((size_t)2 * HELEMS * 2);

  const int ktab[6]  = {2, 3, 4, 2, 4, 8};
  const int wbase[6] = {0, 2, 5, 9, 11, 15};

  // bf16 conversions
  {
    int n = NROWS * DMODEL;
    cvt_f32_bf16<<<(n + 255) / 256, 256, 0, stream>>>(x, xb, n);
    int m = 2 * DMODEL * DMODEL;
    cvt_f32_bf16<<<(m + 255) / 256, 256, 0, stream>>>(s6A,  wA,  m);
    cvt_f32_bf16<<<(m + 255) / 256, 256, 0, stream>>>(s6Bm, wBm, m);
    cvt_f32_bf16<<<(m + 255) / 256, 256, 0, stream>>>(s6C,  wC,  m);
    cvt_f32_bf16<<<(m + 255) / 256, 256, 0, stream>>>(s6sw, wsw, m);
  }
  for (int c = 0; c < 6; ++c) {
    int total = 256 * DMODEL * ktab[c];
    pack_conv<<<(total + 255) / 256, 256, 0, stream>>>(
        cw[c], wconv + (size_t)wbase[c] * 256 * DMODEL, ktab[c], total);
  }

  // conv branches (WMMA) -> raw tfa/cfa
  conv_gemm<<<dim3(NROWS / 16, 6), 128, 0, stream>>>(
      xb, wconv, cb[0], cb[1], cb[2], cb[3], cb[4], cb[5], tfa_raw, cfa_raw);

  // triple LN -> agg
  ln_agg_kernel<<<NROWS, 256, 0, stream>>>(tfa_raw, cfa_raw, tfa_g, tfa_be,
                                           cfa_g, cfa_be, feat_g, feat_be, aggf, aggbf);

  // bidirectional chunked S6 scan (raw conv buffers reused as ys outputs)
  for (int dir = 0; dir < 2; ++dir) {
    const unsigned short* wswd = wsw + (size_t)dir * DMODEL * DMODEL;
    const unsigned short* wAd  = wA  + (size_t)dir * DMODEL * DMODEL;
    const unsigned short* wBmd = wBm + (size_t)dir * DMODEL * DMODEL;
    const unsigned short* wCd  = wC  + (size_t)dir * DMODEL * DMODEL;
    const float* sbd = s6sb + dir * DMODEL;
    const float* dvd = s6Dv + dir * DMODEL;
    float* ysd = dir ? cfa_raw : tfa_raw;

    sel_gemm<<<dim3(NROWS / 16, 3), 128, 0, stream>>>(aggbf, wswd, sbd, sel);

    for (int t = 0; t < CS; ++t) {
      const unsigned short* hin = hbf + (size_t)((t ^ 1) & 1) * HELEMS;
      unsigned short* hout      = hbf + (size_t)(t & 1) * HELEMS;
      s6_stepA<<<dim3(NSTATE / 16, 3), 128, 0, stream>>>(
          aggbf, hin, hout, sel, wAd, wBmd, t, dir, t > 0 ? 1 : 0);
      s6_stepB<<<dim3(NSTATE / 16, 3), 128, 0, stream>>>(
          hout, aggf, wCd, dvd, ysd, t, dir);
    }
  }

  // per-direction LN + final concat LN -> d_out
  final_ln_kernel<<<NROWS, 256, 0, stream>>>(tfa_raw, cfa_raw, s6ng, s6nb,
                                             fing, finb, (float*)d_out);
}
